// Block_25228637896844
// MI455X (gfx1250) — matrix operand order repaired and checked by reading, not compiled, not run on hardware
//
#include <hip/hip_runtime.h>
#include <hip/hip_bf16.h>
#include <cmath>
#include <cstdint>

// ---------------------------------------------------------------------------
// Transformer block on gfx1250: bf16 WMMA GEMMs fed by triple-buffered
// global_load_async_to_lds_b128 (ASYNCcnt, 2 chunks in flight), ds_load_b128
// fragment reads, v_wmma_f32_16x16x32_bf16 math.
// Block tile 128M x 64N x 32K, 4 waves, each wave 32M x 64N (8 accumulators).
// ---------------------------------------------------------------------------

typedef __attribute__((ext_vector_type(16))) __bf16 bf16x16;
typedef __attribute__((ext_vector_type(8)))  __bf16 bf16x8;
typedef __attribute__((ext_vector_type(8)))  float  f32x8;

#define BM 128
#define BN 64
#define BK 32
#define PAD 40  // padded LDS row stride in halfs (80B): conflict-free b128 reads

__device__ __forceinline__ void async_b128(const void* g, uint32_t lds_addr) {
  // GLOBAL_LOAD_ASYNC_TO_LDS_B128: VDST = LDS byte address, VADDR = 64b global.
  asm volatile("global_load_async_to_lds_b128 %0, %1, off"
               :: "v"(lds_addr), "v"((uint64_t)(uintptr_t)g)
               : "memory");
}
__device__ __forceinline__ void wait_async_le6() {
  asm volatile("s_wait_asynccnt 0x6" ::: "memory");
}
__device__ __forceinline__ void wait_async_le0() {
  asm volatile("s_wait_asynccnt 0x0" ::: "memory");
}
__device__ __forceinline__ uint32_t lds_addr_of(const void* p) {
  // Low 32 bits of a generic pointer into LDS are the LDS byte offset.
  return (uint32_t)(uintptr_t)p;
}

// ---------------- WMMA fragment loaders from LDS (contiguous k order, shared by A and B) ---------------
// A 16x32 bf16: lane L -> row = base + L%16; halves h -> K = (L/16)*16 + h
// (16 contiguous halfs = two b128 reads).
__device__ __forceinline__ bf16x16 lds_a_frag(const __bf16* s, int mOff, int lane) {
  const int grp = lane >> 4;
  const __bf16* p = s + (size_t)(mOff + (lane & 15)) * PAD + grp * 16;
  union { bf16x16 v; bf16x8 h[2]; } u;
  u.h[0] = *(const bf16x8*)(p);
  u.h[1] = *(const bf16x8*)(p + 8);
  return u.v;
}
// B 32x16 bf16 from row-major [N,K] tile: lane L -> col = base + L%16;
// halves h -> K = (L/16)*16 + h (16 contiguous halfs = two b128 reads).
__device__ __forceinline__ bf16x16 lds_b_frag(const __bf16* s, int nOff, int lane) {
  const int grp = lane >> 4;
  const __bf16* p = s + (size_t)(nOff + (lane & 15)) * PAD + grp * 16;
  union { bf16x16 v; bf16x8 h[2]; } u;
  u.h[0] = *(const bf16x8*)(p);
  u.h[1] = *(const bf16x8*)(p + 8);
  return u.v;
}

enum { EPI_BF16 = 0, EPI_BF16_GELU = 1, EPI_F32 = 2, EPI_BF16_T = 3 };

template <int EPI>
__global__ __launch_bounds__(128) void gemm_wmma_kernel(
    const __bf16* __restrict__ A, int lda,
    const __bf16* __restrict__ Bt, int ldb,
    void* __restrict__ Cv, int ldc,
    const float* __restrict__ bias,
    const float* __restrict__ resid,
    float scale, int M, int N, int K, int Sdim) {
  __shared__ alignas(16) __bf16 sA[3][BM * PAD];
  __shared__ alignas(16) __bf16 sB[3][BN * PAD];

  const int nBlocks = N / BN;
  const int mBlock  = (int)(blockIdx.x / nBlocks) * BM;
  const int nBlock  = (int)(blockIdx.x % nBlocks) * BN;
  const int tid  = threadIdx.x;
  const int wave = tid >> 5;
  const int lane = tid & 31;

  // Thread-constant copy coordinates: transfer j = tid + t*128 -> row =
  // (tid>>2) + 32*t, seg = tid&3 (both per-thread constants -> no exec mask).
  const int cRow = tid >> 2;
  const int cSeg = (tid & 3) * 8;
  const __bf16* gAi = A  + (size_t)(mBlock + cRow) * lda + cSeg;  // chunk 0
  const __bf16* gBi = Bt + (size_t)(nBlock + cRow) * ldb + cSeg;
  uint32_t lA[3], lB[3];
#pragma unroll
  for (int b = 0; b < 3; ++b) {
    lA[b] = lds_addr_of(&sA[b][(size_t)cRow * PAD + cSeg]);
    lB[b] = lds_addr_of(&sB[b][(size_t)cRow * PAD + cSeg]);
  }

  int issueBuf = 0;
  auto issue = [&]() {
#pragma unroll
    for (int t = 0; t < 4; ++t)  // A tile: 4 x 128 b128 transfers
      async_b128(gAi + (size_t)t * 32 * lda, lA[issueBuf] + t * 32 * PAD * 2);
#pragma unroll
    for (int t = 0; t < 2; ++t)  // B tile: 2 x 128 b128 transfers
      async_b128(gBi + (size_t)t * 32 * ldb, lB[issueBuf] + t * 32 * PAD * 2);
    gAi += BK; gBi += BK;
    issueBuf = (issueBuf == 2) ? 0 : issueBuf + 1;
  };

  f32x8 acc[2][4] = {};
  int cb = 0;  // compute buffer
  auto computeChunk = [&]() {
    const __bf16* sa = sA[cb];
    const __bf16* sb = sB[cb];
    cb = (cb == 2) ? 0 : cb + 1;
    const bf16x16 a0 = lds_a_frag(sa, wave * 32, lane);
    const bf16x16 a1 = lds_a_frag(sa, wave * 32 + 16, lane);
    bf16x16 bfr[4];
#pragma unroll
    for (int t = 0; t < 4; ++t) bfr[t] = lds_b_frag(sb, t * 16, lane);
#pragma unroll
    for (int t = 0; t < 4; ++t) {
      acc[0][t] = __builtin_amdgcn_wmma_f32_16x16x32_bf16(
          false, a0, false, bfr[t], (short)0, acc[0][t], false, false);
      acc[1][t] = __builtin_amdgcn_wmma_f32_16x16x32_bf16(
          false, a1, false, bfr[t], (short)0, acc[1][t], false, false);
    }
  };

  const int nk = K / BK;  // >= 24 for all GEMMs here
  issue();
  issue();

  // Steady state: branch-free wait (next chunk may stay in flight).
  for (int kc = 0; kc < nk - 1; ++kc) {
    wait_async_le6();       // chunk kc landed; chunk kc+1 still in flight
    __syncthreads();        // all waves have chunk kc; prev reads finished
    if (kc + 2 < nk) issue();  // into buffer last read at kc-1 (behind barrier)
    computeChunk();
  }
  // Peeled tail: only the last chunk is outstanding.
  wait_async_le0();
  __syncthreads();
  computeChunk();

  // C/D layout: lane holds col N = lane%16; VGPR r holds row M = r + 8*(lane/16)
  const int nOff = lane & 15;
  const int mOff = (lane >> 4) * 8;
#pragma unroll
  for (int mt = 0; mt < 2; ++mt) {
    const int mBase = mBlock + wave * 32 + mt * 16;
#pragma unroll
    for (int t = 0; t < 4; ++t) {
      const int n = nBlock + 16 * t + nOff;
      const float bn = bias ? bias[n] : 0.0f;
#pragma unroll
      for (int r = 0; r < 8; ++r) {
        const int m = mBase + mOff + r;
        float v = acc[mt][t][r] * scale + bn;
        if (EPI == EPI_BF16_GELU)
          v = 0.5f * v * (1.0f + erff(v * 0.70710678118654752f));  // exact GELU
        if (EPI == EPI_F32) {
          float o = v;
          if (resid) o += resid[(size_t)m * ldc + n];
          ((float*)Cv)[(size_t)m * ldc + n] = o;
        } else if (EPI == EPI_BF16_T) {
          // vT[b][n][s] with b = m/Sdim, s = m%Sdim
          const size_t addr = ((size_t)(m / Sdim) * N + n) * Sdim + (m % Sdim);
          ((__bf16*)Cv)[addr] = (__bf16)v;
        } else {
          ((__bf16*)Cv)[(size_t)m * ldc + n] = (__bf16)v;
        }
      }
    }
  }
}

// ------------------------------ LayerNorm -----------------------------------
__global__ __launch_bounds__(256) void ln_bf16_kernel(
    const float* __restrict__ x, const float* __restrict__ g,
    const float* __restrict__ b, __bf16* __restrict__ out, int D) {
  const int row = blockIdx.x;
  const float* xr = x + (size_t)row * D;
  float s = 0.f, ss = 0.f;
  for (int i = threadIdx.x; i < D; i += blockDim.x) {
    float v = xr[i]; s += v; ss += v * v;
  }
  for (int o = 16; o > 0; o >>= 1) {
    s  += __shfl_xor(s, o, 32);
    ss += __shfl_xor(ss, o, 32);
  }
  __shared__ float shs[8], shss[8];
  const int wid = threadIdx.x >> 5;
  if ((threadIdx.x & 31) == 0) { shs[wid] = s; shss[wid] = ss; }
  __syncthreads();
  float ts = 0.f, tss = 0.f;
  for (int w = 0; w < 8; ++w) { ts += shs[w]; tss += shss[w]; }
  const float mean = ts / D;
  const float var  = tss / D - mean * mean;
  const float rstd = rsqrtf(var + 1e-5f);
  for (int i = threadIdx.x; i < D; i += blockDim.x) {
    float v = (xr[i] - mean) * rstd * g[i] + b[i];
    out[(size_t)row * D + i] = (__bf16)v;
  }
}

// ------------------------------ Softmax -------------------------------------
__global__ __launch_bounds__(256) void softmax_bf16_kernel(
    const float* __restrict__ scores, __bf16* __restrict__ attn, int S) {
  const int row  = blockIdx.x;
  const float* sr = scores + (size_t)row * S;
  const int base = threadIdx.x * 8;
  float loc[8];
  float mx = -3.4e38f;
#pragma unroll
  for (int i = 0; i < 8; ++i) { loc[i] = sr[base + i]; mx = fmaxf(mx, loc[i]); }
  for (int o = 16; o > 0; o >>= 1) mx = fmaxf(mx, __shfl_xor(mx, o, 32));
  __shared__ float shm[8], shsum[8];
  const int wid = threadIdx.x >> 5;
  if ((threadIdx.x & 31) == 0) shm[wid] = mx;
  __syncthreads();
  float m2 = shm[0];
  for (int w = 1; w < 8; ++w) m2 = fmaxf(m2, shm[w]);
  float sum = 0.f;
#pragma unroll
  for (int i = 0; i < 8; ++i) { loc[i] = expf(loc[i] - m2); sum += loc[i]; }
  for (int o = 16; o > 0; o >>= 1) sum += __shfl_xor(sum, o, 32);
  if ((threadIdx.x & 31) == 0) shsum[wid] = sum;
  __syncthreads();
  float tot = 0.f;
  for (int w = 0; w < 8; ++w) tot += shsum[w];
  const float inv = 1.0f / tot;
#pragma unroll
  for (int i = 0; i < 8; ++i)
    attn[(size_t)row * S + base + i] = (__bf16)(loc[i] * inv);
}

// ------------------- fp32 -> bf16 weight convert + transpose ----------------
__global__ __launch_bounds__(256) void convert_transpose_kernel(
    const float* __restrict__ W, __bf16* __restrict__ WT, int K, int N) {
  const size_t idx = (size_t)blockIdx.x * blockDim.x + threadIdx.x;
  if (idx >= (size_t)K * N) return;
  const int k = (int)(idx / N);
  const int n = (int)(idx % N);
  WT[(size_t)n * K + k] = (__bf16)W[idx];
}

// ---------------------------------------------------------------------------
extern "C" void kernel_launch(void* const* d_in, const int* in_sizes, int n_in,
                              void* d_out, int out_size, void* d_ws, size_t ws_size,
                              hipStream_t stream) {
  const int D = 768, H = 3072, S = 2048, Bb = 8, M = Bb * S;

  const float* x     = (const float*)d_in[0];
  const float* ln1_g = (const float*)d_in[1];
  const float* ln1_b = (const float*)d_in[2];
  const float* ln2_g = (const float*)d_in[3];
  const float* ln2_b = (const float*)d_in[4];
  const float* Wq = (const float*)d_in[5];  const float* bq = (const float*)d_in[6];
  const float* Wk = (const float*)d_in[7];  const float* bk = (const float*)d_in[8];
  const float* Wv = (const float*)d_in[9];  const float* bv = (const float*)d_in[10];
  const float* Wo = (const float*)d_in[11]; const float* bo = (const float*)d_in[12];
  const float* Wfc = (const float*)d_in[13]; const float* bfc = (const float*)d_in[14];
  const float* Wpj = (const float*)d_in[15]; const float* bpj = (const float*)d_in[16];

  char* wp = (char*)d_ws;
  auto alloc = [&](size_t bytes) -> void* {
    char* p = wp; wp += (bytes + 255) & ~size_t(255); return (void*)p;
  };
  __bf16* WqT  = (__bf16*)alloc((size_t)D * D * 2);
  __bf16* WkT  = (__bf16*)alloc((size_t)D * D * 2);
  __bf16* WvT  = (__bf16*)alloc((size_t)D * D * 2);
  __bf16* WoT  = (__bf16*)alloc((size_t)D * D * 2);
  __bf16* WfcT = (__bf16*)alloc((size_t)D * H * 2);
  __bf16* WpjT = (__bf16*)alloc((size_t)H * D * 2);
  __bf16* h    = (__bf16*)alloc((size_t)M * D * 2);
  __bf16* qb   = (__bf16*)alloc((size_t)M * D * 2);
  __bf16* kb   = (__bf16*)alloc((size_t)M * D * 2);
  __bf16* vT   = (__bf16*)alloc((size_t)Bb * D * S * 2);
  __bf16* yb   = (__bf16*)alloc((size_t)M * D * 2);
  float*  x1   = (float*) alloc((size_t)M * D * 4);
  __bf16* h2   = (__bf16*)alloc((size_t)M * D * 2);
  __bf16* mb   = (__bf16*)alloc((size_t)M * H * 2);
  float*  sc   = (float*) alloc((size_t)S * S * 4);
  __bf16* at   = (__bf16*)alloc((size_t)S * S * 2);
  (void)ws_size; (void)in_sizes; (void)n_in; (void)out_size;

  auto cvtg = [](size_t total) { return dim3((unsigned)((total + 255) / 256)); };
  auto gemg = [](int m, int n) { return dim3((unsigned)((size_t)(m / BM) * (n / BN))); };
  const dim3 blk128(128), blk256(256);
  const float inv_sqrt_d = 1.0f / sqrtf((float)D);

  // 1) weight conversion + transpose
  convert_transpose_kernel<<<cvtg((size_t)D * D), blk256, 0, stream>>>(Wq, WqT, D, D);
  convert_transpose_kernel<<<cvtg((size_t)D * D), blk256, 0, stream>>>(Wk, WkT, D, D);
  convert_transpose_kernel<<<cvtg((size_t)D * D), blk256, 0, stream>>>(Wv, WvT, D, D);
  convert_transpose_kernel<<<cvtg((size_t)D * D), blk256, 0, stream>>>(Wo, WoT, D, D);
  convert_transpose_kernel<<<cvtg((size_t)D * H), blk256, 0, stream>>>(Wfc, WfcT, D, H);
  convert_transpose_kernel<<<cvtg((size_t)H * D), blk256, 0, stream>>>(Wpj, WpjT, H, D);

  // 2) LN1
  ln_bf16_kernel<<<dim3(M), blk256, 0, stream>>>(x, ln1_g, ln1_b, h, D);

  // 3) QKV projections
  gemm_wmma_kernel<EPI_BF16><<<gemg(M, D), blk128, 0, stream>>>(
      h, D, WqT, D, qb, D, bq, nullptr, 1.0f, M, D, D, S);
  gemm_wmma_kernel<EPI_BF16><<<gemg(M, D), blk128, 0, stream>>>(
      h, D, WkT, D, kb, D, bk, nullptr, 1.0f, M, D, D, S);
  gemm_wmma_kernel<EPI_BF16_T><<<gemg(M, D), blk128, 0, stream>>>(
      h, D, WvT, D, vT, D, bv, nullptr, 1.0f, M, D, D, S);

  // 4) attention, one batch at a time (16MB scores scratch stays L2-resident)
  for (int b = 0; b < Bb; ++b) {
    const __bf16* qp = qb + (size_t)b * S * D;
    const __bf16* kp = kb + (size_t)b * S * D;
    gemm_wmma_kernel<EPI_F32><<<gemg(S, S), blk128, 0, stream>>>(
        qp, D, kp, D, sc, S, nullptr, nullptr, inv_sqrt_d, S, S, D, S);
    softmax_bf16_kernel<<<dim3(S), blk256, 0, stream>>>(sc, at, S);
    gemm_wmma_kernel<EPI_BF16><<<gemg(S, D), blk128, 0, stream>>>(
        at, S, vT + (size_t)b * D * S, S, yb + (size_t)b * S * D, D,
        nullptr, nullptr, 1.0f, S, D, S, S);
  }

  // 5) output projection + residual
  gemm_wmma_kernel<EPI_F32><<<gemg(M, D), blk128, 0, stream>>>(
      yb, D, WoT, D, x1, D, bo, x, 1.0f, M, D, D, S);

  // 6) MLP
  ln_bf16_kernel<<<dim3(M), blk256, 0, stream>>>(x1, ln2_g, ln2_b, h2, D);
  gemm_wmma_kernel<EPI_BF16_GELU><<<gemg(M, H), blk128, 0, stream>>>(
      h2, D, WfcT, D, mb, H, bfc, nullptr, 1.0f, M, H, D, S);
  gemm_wmma_kernel<EPI_F32><<<gemg(M, D), blk128, 0, stream>>>(
      mb, H, WpjT, H, (float*)d_out, D, bpj, x1, 1.0f, M, D, H, S);
}
